// VideoCB_ConvLSTM_19018115187376
// MI455X (gfx1250) — compile-verified, matrix operand order checked by reading, not run on hardware
//
#include <hip/hip_runtime.h>

typedef __bf16 bf16_t;
typedef __bf16 v16bf __attribute__((ext_vector_type(16)));
typedef float  v8f   __attribute__((ext_vector_type(8)));

#define NB   4
#define CX   16
#define DT   24
#define HH   104
#define WW   150
#define HID  32
#define HWI  (HH * WW)        // 15600
#define MTOT (NB * HWI)       // 62400
#define COUT 128
#define KTOT 1200             // 25 taps * 48 channels (16 x + 32 h), no per-tap pad
#define NCHUNK 38             // ceil(1200/32): 12 tap-pair periods (3 chunks) + 2 tail
#define STATE_N (NB * HID * HWI)
#define XC_N ((size_t)DT * NB * HWI * CX)
#define WSW_DWORDS (NCHUNK * 8 * 32 * 8)   // [chunk][nt][lane][dword] = 77824

__device__ __forceinline__ float sigmoid_f(float x) {
    return 1.0f / (1.0f + __expf(-x));
}
__device__ __forceinline__ float tanh_f(float x) {
    float ax = fabsf(x);
    float e  = __expf(-2.0f * ax);
    float r  = (1.0f - e) / (1.0f + e);
    return copysignf(r, x);
}

// ---------------------------------------------------------------------------
// x (N,C,D,H,W) f32 -> xc bf16, layout [t][n][pix][c16]  (channel-minor)
// ---------------------------------------------------------------------------
__global__ void convert_x(const float* __restrict__ x, bf16_t* __restrict__ xc) {
    size_t idx = (size_t)blockIdx.x * 256 + threadIdx.x;
    if (idx >= XC_N) return;
    int c    = (int)(idx & 15);
    size_t p = idx >> 4;
    int pix  = (int)(p % HWI);  p /= HWI;
    int n    = (int)(p % NB);
    int tt   = (int)(p / NB);
    xc[idx] = (bf16_t)x[(((size_t)n * CX + c) * DT + tt) * HWI + pix];
}

// ---------------------------------------------------------------------------
// Wconv (128,48,5,5) f32 -> fragment-swizzled bf16 pairs:
// Wsw[chunk][nt][lane][r] = packed {w(n,k), w(n,k+1)} with
//   n = nt*16 + (lane&15),  k = chunk*32 + klocal(lane, 2r),
//   klocal(e) = (e<8 ? e : e+8) + (lane>=16 ? 8 : 0)   (B fragment layout)
//   k = tap*48 + c flat (c: 0..15 = x ch, 16..47 = h ch); k >= 1200 -> zero
// ---------------------------------------------------------------------------
__global__ void convert_wsw(const float* __restrict__ Wc, uint32_t* __restrict__ Wsw) {
    int idx = blockIdx.x * 256 + threadIdx.x;   // dword index
    if (idx >= WSW_DWORDS) return;
    int r     = idx & 7;
    int lane  = (idx >> 3) & 31;
    int nt    = (idx >> 8) & 7;
    int chunk = idx >> 11;
    int n = nt * 16 + (lane & 15);
    int e = 2 * r;
    int klocal = (e < 8 ? e : e + 8) + (lane >= 16 ? 8 : 0);
    int k   = chunk * 32 + klocal;              // even; pair (k, k+1) shares a tap
    int tap = k / 48;
    int c   = k - tap * 48;
    float f0 = (k     < KTOT) ? Wc[((size_t)n * 48 + c    ) * 25 + tap] : 0.0f;
    float f1 = (k + 1 < KTOT) ? Wc[((size_t)n * 48 + c + 1) * 25 + tap] : 0.0f;
    union { bf16_t b[2]; uint32_t u; } pk;
    pk.b[0] = (bf16_t)f0;
    pk.b[1] = (bf16_t)f1;
    Wsw[idx] = pk.u;
}

__global__ void init_state(float* __restrict__ c, bf16_t* __restrict__ h) {
    int idx = blockIdx.x * 256 + threadIdx.x;
    if (idx >= STATE_N) return;
    c[idx] = 0.0f;
    h[idx] = (bf16_t)0.0f;
}

// ---------------------------------------------------------------------------
// One ConvLSTM timestep. Workgroup = 256 threads = 8 waves, M-tile = 256
// positions (2 subtiles of 16 per wave). Each wave accumulates all 128 gate
// channels: acc[2][8] v8f. Both operand streams are software-pipelined one
// full chunk ahead: B chunks double-buffered in LDS (registers stored after
// each barrier were fetched a chunk earlier), and A fragments for chunk ch+1
// are issued before mma_chunk(ch) so their L2 latency hides under the WMMA
// stream. K is packed (no per-tap padding): a 2-tap group = exactly 3 chunks
// of 32, each chunk = two 16B channel-minor loads per subtile.
// ---------------------------------------------------------------------------
__global__ __launch_bounds__(256)
void convlstm_step(const bf16_t* __restrict__ xc, const float* __restrict__ bias,
                   const uint32_t* __restrict__ Wsw,
                   const bf16_t* __restrict__ hin, bf16_t* __restrict__ hout,
                   float* __restrict__ cbuf, float* __restrict__ out, int t)
{
    __shared__ __align__(16) uint32_t blds[2][2048];   // 2 x 8 KB B chunk buffers

    const int  tid    = threadIdx.x;
    const int  lane   = tid & 31;
    const int  wave   = tid >> 5;
    const int  mbase  = blockIdx.x * 256;
    const bool hiHalf = lane >= 16;
    const int  hio    = hiHalf ? 8 : 0;

    // --- per-subtile A row position (A layout: both lane halves hold M=0..15)
    int yA[2], xA[2];
    const bf16_t* xb[2];    // x  pixel base + channel offset (c hio..hio+7)
    const bf16_t* hb0[2];   // h  pixel base + channels hio..hio+7
    const bf16_t* hb16[2];  // h  pixel base + channels 16+hio..23+hio
    #pragma unroll
    for (int s = 0; s < 2; ++s) {
        int m  = mbase + (wave * 2 + s) * 16 + (lane & 15);
        int mc = m < MTOT ? m : MTOT - 1;        // clamp: EXEC stays all-ones
        int n   = mc / HWI;
        int rem = mc - n * HWI;
        int y   = rem / WW;
        yA[s] = y;
        xA[s] = rem - y * WW;
        xb[s]   = xc  + ((size_t)(t * NB + n) * HWI) * CX + hio;
        hb0[s]  = hin + (size_t)n * HWI * HID + hio;
        hb16[s] = hb0[s] + 16;
    }

    // replicate-clamped pixel index for a tap
    auto mksp = [&](int tap, int s) -> int {
        int ky = tap / 5;
        int kx = tap - ky * 5;
        int sy = yA[s] + ky - 2; sy = sy < 0 ? 0 : (sy > HH - 1 ? HH - 1 : sy);
        int sx = xA[s] + kx - 2; sx = sx < 0 ? 0 : (sx > WW - 1 ? WW - 1 : sx);
        return sy * WW + sx;
    };

    // B loader: thread (wave, lane) owns fragment slot (nt = wave, lane)
    const int wslot4 = (wave * 32 + lane) * 2;   // uint4 index within a chunk (512/chunk)
    const uint4* wsw4 = (const uint4*)Wsw;
    const int ldsslot = (wave * 32 + lane) * 8;  // dword index of this thread's slot

    v8f acc[2][8] = {};

    // ---- 2-deep B pipeline prologue ----
    // stage B(0) into buf0; leave B(1) resident in nb registers
    uint4 nb0, nb1;
    {
        uint4 b0 = wsw4[wslot4 + 0];
        uint4 b1 = wsw4[wslot4 + 1];
        uint4* d = (uint4*)&blds[0][ldsslot];
        d[0] = b0; d[1] = b1;
        const uint4* src = wsw4 + (size_t)512 + wslot4;   // chunk 1
        nb0 = src[0];
        nb1 = src[1];
    }

    union AF { v16bf bf; uint4 q[2]; };
    int p = 0;   // LDS buffer parity holding the current chunk

    // barrier; ds_store resident B(ch+1); 16 WMMAs on buf[p]; fetch B(ch+2)
    auto mma_chunk = [&](int ch, const AF* af) {
        __syncthreads();
        if (ch + 1 < NCHUNK) {
            uint4* d = (uint4*)&blds[p ^ 1][ldsslot];
            d[0] = nb0; d[1] = nb1;     // data already in registers: no load-wait
        }
        const uint32_t* bufp = blds[p];
        #pragma unroll
        for (int nt = 0; nt < 8; ++nt) {
            union BF { v16bf bf; uint4 q[2]; } b;
            const uint4* s4 = (const uint4*)&bufp[(nt * 32 + lane) * 8];
            b.q[0] = s4[0];
            b.q[1] = s4[1];
            acc[0][nt] = __builtin_amdgcn_wmma_f32_16x16x32_bf16(
                false, af[0].bf, false, b.bf, (short)0, acc[0][nt], false, false);
            acc[1][nt] = __builtin_amdgcn_wmma_f32_16x16x32_bf16(
                false, af[1].bf, false, b.bf, (short)0, acc[1][nt], false, false);
        }
        if (ch + 2 < NCHUNK) {          // refill pipeline; latency spans next barrier
            const uint4* src = wsw4 + (size_t)(ch + 2) * 512 + wslot4;
            nb0 = src[0];
            nb1 = src[1];
        }
        p ^= 1;
    };

    // A chunk loaders (channel-minor: one 16B load per 8 fragment elements)
    auto loadT0 = [&](AF* af, const int* sp) {       // x[sp] | h[sp, 0..15]
        #pragma unroll
        for (int s = 0; s < 2; ++s) {
            af[s].q[0] = *(const uint4*)(xb[s]  + (size_t)sp[s] * CX);
            af[s].q[1] = *(const uint4*)(hb0[s] + (size_t)sp[s] * HID);
        }
    };
    auto loadT1 = [&](AF* af, const int* spa, const int* spb) {  // h[spa,16..31] | x[spb]
        #pragma unroll
        for (int s = 0; s < 2; ++s) {
            af[s].q[0] = *(const uint4*)(hb16[s] + (size_t)spa[s] * HID);
            af[s].q[1] = *(const uint4*)(xb[s]   + (size_t)spb[s] * CX);
        }
    };
    auto loadT2 = [&](AF* af, const int* sp) {       // h[sp,0..15] | h[sp,16..31]
        #pragma unroll
        for (int s = 0; s < 2; ++s) {
            af[s].q[0] = *(const uint4*)(hb0[s]  + (size_t)sp[s] * HID);
            af[s].q[1] = *(const uint4*)(hb16[s] + (size_t)sp[s] * HID);
        }
    };

    // ---- A pipeline prologue: sp for period 0, load chunk 0 fragments ----
    int spA[2], spB[2];
    #pragma unroll
    for (int s = 0; s < 2; ++s) {
        spA[s] = mksp(0, s);
        spB[s] = mksp(1, s);
    }
    AF af0[2], af1[2], af2[2];
    loadT0(af0, spA);

    // ---- 12 tap-pair periods (3 packed chunks each); A loaded 1 chunk ahead
    for (int per = 0; per < 12; ++per) {
        const int ch = 3 * per;
        loadT1(af1, spA, spB);       // chunk ch+1 fragments (ahead)
        mma_chunk(ch, af0);
        loadT2(af2, spB);            // chunk ch+2 fragments (ahead)
        mma_chunk(ch + 1, af1);
        // next period's taps (per=11 -> tap 24 = tail; tap 25 sp is unused but safe)
        int spA2[2], spB2[2];
        const int tapN = 2 * per + 2;
        #pragma unroll
        for (int s = 0; s < 2; ++s) {
            spA2[s] = mksp(tapN, s);
            spB2[s] = mksp(tapN + 1, s);
        }
        loadT0(af0, spA2);           // first chunk of next period / tail chunk 36
        mma_chunk(ch + 2, af2);
        #pragma unroll
        for (int s = 0; s < 2; ++s) { spA[s] = spA2[s]; spB[s] = spB2[s]; }
    }

    // ---- tail: tap 24 (1.5 chunks; af0 already holds chunk 36 fragments) ----
    {
        AF af1t[2];
        #pragma unroll
        for (int s = 0; s < 2; ++s) {   // chunk 37: tap24 c32..47 | zero pad
            af1t[s].q[0] = *(const uint4*)(hb16[s] + (size_t)spA[s] * HID);
            af1t[s].q[1] = make_uint4(0u, 0u, 0u, 0u);
        }
        mma_chunk(36, af0);
        mma_chunk(37, af1t);
    }

    // ---- fused bias + LSTM gate math.
    // C/D layout: VGPR r -> M = r (+8 for hi lanes), lane&15 -> N column.
    // Gates: i = nt 0,1   f = nt 2,3   g = nt 4,5   o = nt 6,7.
    const int hcl = lane & 15;
    float bv[8];
    #pragma unroll
    for (int nt = 0; nt < 8; ++nt) bv[nt] = bias[nt * 16 + hcl];

    #pragma unroll
    for (int s = 0; s < 2; ++s) {
        const int m0 = mbase + (wave * 2 + s) * 16 + (hiHalf ? 8 : 0);
        #pragma unroll
        for (int r = 0; r < 8; ++r) {
            const int m = m0 + r;
            if (m < MTOT) {
                const int nb  = m / HWI;
                const int rem = m - nb * HWI;
                #pragma unroll
                for (int b = 0; b < 2; ++b) {
                    const int hc = b * 16 + hcl;
                    float iv = acc[s][0 + b][r] + bv[0 + b];
                    float fv = acc[s][2 + b][r] + bv[2 + b];
                    float gv = acc[s][4 + b][r] + bv[4 + b];
                    float ov = acc[s][6 + b][r] + bv[6 + b];
                    const size_t sidx = ((size_t)nb * HWI + rem) * HID + hc;  // channel-minor
                    float cold = cbuf[sidx];
                    float cn   = sigmoid_f(fv) * cold + sigmoid_f(iv) * tanh_f(gv);
                    float hn   = sigmoid_f(ov) * tanh_f(cn);
                    cbuf[sidx] = cn;
                    hout[sidx] = (bf16_t)hn;   // bf16 state feeds next step's A matrix
                    out[(((size_t)nb * HID + hc) * DT + t) * HWI + rem] = hn;
                }
            }
        }
    }
}

// ---------------------------------------------------------------------------
extern "C" void kernel_launch(void* const* d_in, const int* in_sizes, int n_in,
                              void* d_out, int out_size, void* d_ws, size_t ws_size,
                              hipStream_t stream) {
    const float* x    = (const float*)d_in[0];   // (4,16,24,104,150) f32
    const float* Wc   = (const float*)d_in[1];   // (128,48,5,5) f32
    const float* bias = (const float*)d_in[2];   // (128,) f32
    float* out = (float*)d_out;                  // (4,32,24,104,150) f32

    char* ws = (char*)d_ws;
    size_t off = 0;
    uint32_t* Wsw = (uint32_t*)(ws + off);
    off += (size_t)WSW_DWORDS * 4;                          //   311,296 B
    off = (off + 255) & ~(size_t)255;
    bf16_t* xc = (bf16_t*)(ws + off);
    off += XC_N * sizeof(bf16_t);                           // 47,923,200 B
    off = (off + 255) & ~(size_t)255;
    float* cbuf = (float*)(ws + off);
    off += (size_t)STATE_N * sizeof(float);                 //  7,987,200 B
    off = (off + 255) & ~(size_t)255;
    bf16_t* h0 = (bf16_t*)(ws + off);
    off += (size_t)STATE_N * sizeof(bf16_t);                //  3,993,600 B
    off = (off + 255) & ~(size_t)255;
    bf16_t* h1 = (bf16_t*)(ws + off);                       //  3,993,600 B (~64.2 MB total)

    convert_wsw<<<(WSW_DWORDS + 255) / 256, 256, 0, stream>>>(Wc, Wsw);
    convert_x<<<(int)((XC_N + 255) / 256), 256, 0, stream>>>(x, xc);
    init_state<<<(STATE_N + 255) / 256, 256, 0, stream>>>(cbuf, h0);

    const int blocks = (MTOT + 255) / 256;     // 244 workgroups per timestep
    for (int tstep = 0; tstep < DT; ++tstep) {
        const bf16_t* hin  = (tstep & 1) ? h1 : h0;
        bf16_t*       hnew = (tstep & 1) ? h0 : h1;
        convlstm_step<<<blocks, 256, 0, stream>>>(xc, bias, Wsw, hin, hnew,
                                                  cbuf, out, tstep);
    }
}
